// Attention_59210419143335
// MI455X (gfx1250) — matrix operand order repaired and checked by reading, not compiled, not run on hardware
//
#include <hip/hip_runtime.h>

// ---------------------------------------------------------------------------
// CDNA5 (gfx1250) axial attention: QKV GEMM -> time attn -> freq attn -> proj
// All matmuls use v_wmma_f32_16x16x32_bf16 (wave32). GEMMs are double-buffered
// and software-pipelined; global_prefetch_b8 used to warm GL2.
// ---------------------------------------------------------------------------

typedef __bf16 bf16_t;
typedef __bf16 v16bf __attribute__((ext_vector_type(16)));
typedef __bf16 v8bf  __attribute__((ext_vector_type(8)));
typedef float  v8f   __attribute__((ext_vector_type(8)));

#define B_SZ    4
#define T_LEN   256
#define F_LEN   64
#define HEADS   8
#define HD      32
#define DMODEL  256
#define QKV_LD  1280          // 5*DMODEL
#define M_ROWS  65536         // B*T*F
#define INV_SQRT_HD 0.17677669529663687f

__device__ __forceinline__ unsigned short f2bf_bits(float f) {
    unsigned u = __builtin_bit_cast(unsigned, f);
    unsigned r = u + 0x7FFFu + ((u >> 16) & 1u);   // round-to-nearest-even
    return (unsigned short)(r >> 16);
}
__device__ __forceinline__ float bf2f(unsigned short s) {
    unsigned u = ((unsigned)s) << 16;
    return __builtin_bit_cast(float, u);
}

__device__ __forceinline__ v8f v8f_zero() {
    v8f z;
#pragma unroll
    for (int i = 0; i < 8; ++i) z[i] = 0.f;
    return z;
}

// A-matrix fragment (16xK slice, K pattern: k = (i/8)*16 + half*8 + i%8)
// lds is row-major [m][k]; ld (elements) multiple of 8 and row stride in
// bytes a multiple of 16 (ld in {40,48,72} qualify) -> two ds_load_b128.
__device__ __forceinline__ v16bf frag_A(const unsigned short* lds, int m0, int ld) {
    int lane = threadIdx.x & 31;
    int hf   = lane >> 4;
    const unsigned short* p = lds + (size_t)(m0 + (lane & 15)) * ld + hf * 8;
    uint4 lo = *(const uint4*)(p);        // k = hf*8 + 0..7
    uint4 hi = *(const uint4*)(p + 16);   // k = 16 + hf*8 + 0..7
    v8bf a = __builtin_bit_cast(v8bf, lo);
    v8bf b = __builtin_bit_cast(v8bf, hi);
    v16bf r;
#pragma unroll
    for (int i = 0; i < 8; ++i) { r[i] = a[i]; r[i + 8] = b[i]; }
    return r;
}

// B-matrix fragment: lane holds column n, element i -> k = (i/8)*16 + half*8 + i%8.
// ldsT is N-major [n][k] so each lane's 16 k's form two contiguous runs of 8.
__device__ __forceinline__ v16bf frag_B(const unsigned short* ldsT, int n0, int ld) {
    int lane = threadIdx.x & 31;
    int hf   = lane >> 4;
    const unsigned short* p = ldsT + (size_t)(n0 + (lane & 15)) * ld + hf * 8;
    uint4 lo = *(const uint4*)(p);        // i = 0..7
    uint4 hi = *(const uint4*)(p + 16);   // i = 8..15
    v8bf a = __builtin_bit_cast(v8bf, lo);
    v8bf b = __builtin_bit_cast(v8bf, hi);
    v16bf r;
#pragma unroll
    for (int i = 0; i < 8; ++i) { r[i] = a[i]; r[i + 8] = b[i]; }
    return r;
}

__device__ __forceinline__ v8f wmma_bf16(v16bf a, v16bf b, v8f c) {
    return __builtin_amdgcn_wmma_f32_16x16x32_bf16(false, a, false, b,
                                                   (short)0, c, false, false);
}

__device__ __forceinline__ float half_max(float v) {
    v = fmaxf(v, __shfl_xor(v, 1, 32));
    v = fmaxf(v, __shfl_xor(v, 2, 32));
    v = fmaxf(v, __shfl_xor(v, 4, 32));
    v = fmaxf(v, __shfl_xor(v, 8, 32));
    return v;
}
__device__ __forceinline__ float half_sum(float v) {
    v += __shfl_xor(v, 1, 32);
    v += __shfl_xor(v, 2, 32);
    v += __shfl_xor(v, 4, 32);
    v += __shfl_xor(v, 8, 32);
    return v;
}

// ---------------------------------------------------------------------------
// Generic bf16-WMMA GEMM: C[M,N] = A[M,K] * B[K,N]
// Block tile 128x128, 8 waves (4 over M x 2 over N), wave tile 32x64.
// Double-buffered LDS + register pipeline + GL2 prefetch.
// ---------------------------------------------------------------------------
template<bool A_BF16, bool OUT_BF16>
__global__ __launch_bounds__(256)
void gemm_kernel(const void* __restrict__ Aptr, const float* __restrict__ B,
                 void* __restrict__ Cptr, int M, int N, int K)
{
    __shared__ unsigned short As [2][128 * 48];   // [m][k], ld 48 (96B rows)
    __shared__ unsigned short BsT[2][128 * 48];   // [n][k], ld 48

    const int bm = blockIdx.x, bn = blockIdx.y;
    const int tid = threadIdx.x, lane = tid & 31, wid = tid >> 5;
    const int wm = wid & 3, wn = wid >> 2;
    const int hf = lane >> 4;

    const float*          Af = (const float*)Aptr;
    const unsigned short* Ab = (const unsigned short*)Aptr;

    // per-thread staging coordinates (4 slots each for A and B tiles)
    int arow[4], acol[4], brow[4], bcol[4];
#pragma unroll
    for (int it = 0; it < 4; ++it) {
        int s = tid + it * 256;
        arow[it] = s >> 3;  acol[it] = (s & 7) * 4;   // A: 128 rows x 32 cols
        brow[it] = s >> 5;  bcol[it] = (s & 31) * 4;  // B: 32 rows x 128 cols
    }

    float4 areg[4]; uint2 abreg[4]; float4 breg[4];

    auto load_tile = [&](int k0) {
#pragma unroll
        for (int it = 0; it < 4; ++it) {
            if constexpr (A_BF16)
                abreg[it] = *(const uint2*)(Ab + (size_t)(bm * 128 + arow[it]) * K + k0 + acol[it]);
            else
                areg[it] = *(const float4*)(Af + (size_t)(bm * 128 + arow[it]) * K + k0 + acol[it]);
            breg[it] = *(const float4*)(B + (size_t)(k0 + brow[it]) * N + bn * 128 + bcol[it]);
        }
    };
    auto store_tile = [&](int buf) {
#pragma unroll
        for (int it = 0; it < 4; ++it) {
            if constexpr (A_BF16) {
                *(uint2*)(&As[buf][arow[it] * 48 + acol[it]]) = abreg[it];
            } else {
                uint2 pk;
                pk.x = (unsigned)f2bf_bits(areg[it].x) | ((unsigned)f2bf_bits(areg[it].y) << 16);
                pk.y = (unsigned)f2bf_bits(areg[it].z) | ((unsigned)f2bf_bits(areg[it].w) << 16);
                *(uint2*)(&As[buf][arow[it] * 48 + acol[it]]) = pk;
            }
            BsT[buf][(bcol[it] + 0) * 48 + brow[it]] = f2bf_bits(breg[it].x);
            BsT[buf][(bcol[it] + 1) * 48 + brow[it]] = f2bf_bits(breg[it].y);
            BsT[buf][(bcol[it] + 2) * 48 + brow[it]] = f2bf_bits(breg[it].z);
            BsT[buf][(bcol[it] + 3) * 48 + brow[it]] = f2bf_bits(breg[it].w);
        }
    };

    v8f acc[2][4];
#pragma unroll
    for (int mi = 0; mi < 2; ++mi)
#pragma unroll
        for (int ni = 0; ni < 4; ++ni) acc[mi][ni] = v8f_zero();

    const int NK = K / 32;
    load_tile(0);
    int buf = 0;
    for (int kt = 0; kt < NK; ++kt) {
        store_tile(buf);
        __syncthreads();
        if (kt + 1 < NK) {
            load_tile((kt + 1) * 32);            // overlap with WMMA below
            if (kt + 2 < NK) {                   // warm GL2 for tile after next
                int k2 = (kt + 2) * 32;
                const void* pa = A_BF16
                    ? (const void*)(Ab + (size_t)(bm * 128 + arow[0]) * K + k2)
                    : (const void*)(Af + (size_t)(bm * 128 + arow[0]) * K + k2);
                __builtin_prefetch(pa, 0, 1);
                __builtin_prefetch(B + (size_t)(k2 + brow[0]) * N + bn * 128 + bcol[0], 0, 1);
            }
        }
        // ---- 8 WMMAs on the staged tile ----
        v16bf af[2];
#pragma unroll
        for (int mi = 0; mi < 2; ++mi) af[mi] = frag_A(As[buf], wm * 32 + mi * 16, 48);
        v16bf bfv[4];
#pragma unroll
        for (int ni = 0; ni < 4; ++ni) bfv[ni] = frag_B(BsT[buf], wn * 64 + ni * 16, 48);
#pragma unroll
        for (int mi = 0; mi < 2; ++mi)
#pragma unroll
            for (int ni = 0; ni < 4; ++ni)
                acc[mi][ni] = wmma_bf16(af[mi], bfv[ni], acc[mi][ni]);
        buf ^= 1;
    }

    // ---- epilogue (C layout: lane n = col, VGPR r -> row hf*8+r) ----
#pragma unroll
    for (int mi = 0; mi < 2; ++mi)
#pragma unroll
        for (int ni = 0; ni < 4; ++ni)
#pragma unroll
            for (int r = 0; r < 8; ++r) {
                int row = bm * 128 + wm * 32 + mi * 16 + hf * 8 + r;
                int col = bn * 128 + wn * 64 + ni * 16 + (lane & 15);
                if constexpr (OUT_BF16)
                    ((unsigned short*)Cptr)[(size_t)row * N + col] = f2bf_bits(acc[mi][ni][r]);
                else
                    ((float*)Cptr)[(size_t)row * N + col] = acc[mi][ni][r];
            }
}

// ---------------------------------------------------------------------------
// Time attention: one workgroup per (b,f,h). 8 waves x 32 query rows = T=256.
// Flash-style online softmax over 8 key tiles of 32. Rotary applied on stage.
// Output v_t -> vt[(((b*F+f)*H+h)*T + t)*HD + d]  (bf16)
// ---------------------------------------------------------------------------
__global__ __launch_bounds__(256)
void time_attn_kernel(const unsigned short* __restrict__ qkv,
                      const float* __restrict__ rot_t,
                      unsigned short* __restrict__ vt)
{
    __shared__ unsigned short qs  [T_LEN * 40];    // [t][hd]  (rotary+scaled q)
    __shared__ unsigned short ks  [32 * 40];       // [key][hd] (rotary k)
    __shared__ unsigned short vst [32 * 40];       // [hd][key] (v transposed)
    __shared__ unsigned short pbuf[8 * 32 * 40];   // per-wave P (32x32)

    const int bfh = blockIdx.x;
    const int h = bfh % HEADS, bf = bfh / HEADS;
    const int f = bf % F_LEN,  b  = bf / F_LEN;

    const size_t rstride = (size_t)F_LEN * QKV_LD;            // stride over t
    const unsigned short* qbase = qkv + ((size_t)b * T_LEN * F_LEN + f) * QKV_LD + h * HD;
    const unsigned short* kbase = qbase + 2 * DMODEL;
    const unsigned short* vbase = qbase + 4 * DMODEL;

    const int tid = threadIdx.x;
    const int lane = tid & 31, wid = tid >> 5, hf = lane >> 4;

    // ---- stage all 256 q rows: rotary + 1/sqrt(hd) scale, one row/thread ----
    {
        const int t = tid;
        const unsigned short* qr = qbase + (size_t)t * rstride;
        const float* ang = rot_t + t * HD;
#pragma unroll
        for (int j = 0; j < HD; j += 2) {
            float x0 = bf2f(qr[j]), x1 = bf2f(qr[j + 1]);
            float a0 = ang[j], a1 = ang[j + 1];
            float o0 = x0 * __cosf(a0) - x1 * __sinf(a0);
            float o1 = x1 * __cosf(a1) + x0 * __sinf(a1);
            qs[t * 40 + j]     = f2bf_bits(o0 * INV_SQRT_HD);
            qs[t * 40 + j + 1] = f2bf_bits(o1 * INV_SQRT_HD);
        }
    }
    __syncthreads();

    v16bf qa[2];
    qa[0] = frag_A(qs, wid * 32, 40);
    qa[1] = frag_A(qs, wid * 32 + 16, 40);

    v8f accO[2][2];
    float rowM[2][8], rowL[2][8];
#pragma unroll
    for (int mi = 0; mi < 2; ++mi) {
#pragma unroll
        for (int ni = 0; ni < 2; ++ni) accO[mi][ni] = v8f_zero();
#pragma unroll
        for (int r = 0; r < 8; ++r) { rowM[mi][r] = -__builtin_inff(); rowL[mi][r] = 0.f; }
    }

    for (int kt = 0; kt < 8; ++kt) {
        if (kt + 1 < 8) {   // warm GL2 for next K/V tile
            int tp = (kt + 1) * 32 + (tid & 31);
            __builtin_prefetch(kbase + (size_t)tp * rstride, 0, 1);
            __builtin_prefetch(vbase + (size_t)tp * rstride, 0, 1);
        }
        __syncthreads();   // previous tile fully consumed
        // ---- stage k tile (rotary) + v tile (transposed), 512 pair jobs ----
#pragma unroll
        for (int it = 0; it < 2; ++it) {
            int s  = tid + it * 256;
            int kr = s >> 4;               // key row in tile
            int j  = (s & 15) * 2;         // dim pair
            int t  = kt * 32 + kr;
            const unsigned short* krp = kbase + (size_t)t * rstride;
            const float* ang = rot_t + t * HD;
            float x0 = bf2f(krp[j]), x1 = bf2f(krp[j + 1]);
            float a0 = ang[j], a1 = ang[j + 1];
            ks[kr * 40 + j]     = f2bf_bits(x0 * __cosf(a0) - x1 * __sinf(a0));
            ks[kr * 40 + j + 1] = f2bf_bits(x1 * __cosf(a1) + x0 * __sinf(a1));
            const unsigned short* vrp = vbase + (size_t)t * rstride;
            vst[j * 40 + kr]       = vrp[j];
            vst[(j + 1) * 40 + kr] = vrp[j + 1];
        }
        __syncthreads();

        // ---- scores S = q @ k^T (already scaled) ----
        v16bf kb0 = frag_B(ks, 0, 40), kb1 = frag_B(ks, 16, 40);
        v8f sc[2][2];
#pragma unroll
        for (int mi = 0; mi < 2; ++mi) {
            sc[mi][0] = wmma_bf16(qa[mi], kb0, v8f_zero());
            sc[mi][1] = wmma_bf16(qa[mi], kb1, v8f_zero());
        }

        // ---- online softmax (rows m = mi*16 + hf*8 + r) ----
#pragma unroll
        for (int mi = 0; mi < 2; ++mi)
#pragma unroll
            for (int r = 0; r < 8; ++r) {
                float v0 = sc[mi][0][r], v1 = sc[mi][1][r];
                float mx = half_max(fmaxf(v0, v1));
                float Mo = rowM[mi][r];
                float Mn = fmaxf(Mo, mx);
                float corr = __expf(Mo - Mn);
                float p0 = __expf(v0 - Mn), p1 = __expf(v1 - Mn);
                float rs = half_sum(p0 + p1);
                rowL[mi][r] = rowL[mi][r] * corr + rs;
                rowM[mi][r] = Mn;
                accO[mi][0][r] *= corr;
                accO[mi][1][r] *= corr;
                sc[mi][0][r] = p0;
                sc[mi][1][r] = p1;
            }

        // ---- relayout P (C-layout -> A-layout) through per-wave LDS ----
        unsigned short* pw = pbuf + wid * 32 * 40;
#pragma unroll
        for (int mi = 0; mi < 2; ++mi)
#pragma unroll
            for (int ni = 0; ni < 2; ++ni)
#pragma unroll
                for (int r = 0; r < 8; ++r)
                    pw[(mi * 16 + hf * 8 + r) * 40 + ni * 16 + (lane & 15)] =
                        f2bf_bits(sc[mi][ni][r]);
        __syncthreads();

        // ---- O += P @ V ----
        v16bf vb0 = frag_B(vst, 0, 40), vb1 = frag_B(vst, 16, 40);
#pragma unroll
        for (int mi = 0; mi < 2; ++mi) {
            v16bf pa = frag_A(pw, mi * 16, 40);
            accO[mi][0] = wmma_bf16(pa, vb0, accO[mi][0]);
            accO[mi][1] = wmma_bf16(pa, vb1, accO[mi][1]);
        }
    }

    // ---- normalize + write vt bf16 ----
    const size_t obase = ((((size_t)b * F_LEN + f) * HEADS + h) * T_LEN) * HD;
#pragma unroll
    for (int mi = 0; mi < 2; ++mi)
#pragma unroll
        for (int ni = 0; ni < 2; ++ni)
#pragma unroll
            for (int r = 0; r < 8; ++r) {
                int t = wid * 32 + mi * 16 + hf * 8 + r;
                int d = ni * 16 + (lane & 15);
                vt[obase + (size_t)t * HD + d] =
                    f2bf_bits(accO[mi][ni][r] / rowL[mi][r]);
            }
}

// ---------------------------------------------------------------------------
// Frequency attention: one workgroup per (b,t,h). 4 waves x 16 rows = F=64.
// Full 64-key softmax. Output y -> ybf[((b*T+t)*F+f)*DMODEL + h*HD + d] (bf16)
// ---------------------------------------------------------------------------
__global__ __launch_bounds__(128)
void freq_attn_kernel(const unsigned short* __restrict__ qkv,
                      const float* __restrict__ rot_f,
                      const unsigned short* __restrict__ vt,
                      unsigned short* __restrict__ ybf)
{
    __shared__ unsigned short qsf [64 * 40];    // [f][hd]
    __shared__ unsigned short ksf [64 * 40];    // [f][hd]
    __shared__ unsigned short vstf[32 * 72];    // [hd][f]
    __shared__ unsigned short pb  [4 * 16 * 72];// per-wave P (16x64), ld 72

    const int bth = blockIdx.x;
    const int h = bth % HEADS, bt = bth / HEADS;
    const int t = bt % T_LEN,  b  = bt / T_LEN;

    const unsigned short* qbase = qkv + (((size_t)b * T_LEN + t) * F_LEN) * QKV_LD
                                      + 1 * DMODEL + h * HD;
    const unsigned short* kbase = qbase + 2 * DMODEL;   // slot 3 (k_f)
    const unsigned short* vtbase = vt + (((size_t)b * F_LEN) * HEADS + h) * T_LEN * HD
                                      + (size_t)t * HD;
    const size_t vstride = (size_t)HEADS * T_LEN * HD;  // stride over f in vt

    const int tid = threadIdx.x;
    const int lane = tid & 31, wid = tid >> 5, hf = lane >> 4;

    // ---- stage q,k (rotary) and v (transposed): 1024 pair jobs / 128 thr ----
#pragma unroll
    for (int it = 0; it < 8; ++it) {
        int s  = tid + it * 128;
        int fr = s >> 4;
        int j  = (s & 15) * 2;
        const float* ang = rot_f + fr * HD;
        float a0 = ang[j], a1 = ang[j + 1];
        float c0 = __cosf(a0), s0 = __sinf(a0);
        float c1 = __cosf(a1), s1 = __sinf(a1);
        const unsigned short* qr = qbase + (size_t)fr * QKV_LD;
        float x0 = bf2f(qr[j]), x1 = bf2f(qr[j + 1]);
        qsf[fr * 40 + j]     = f2bf_bits((x0 * c0 - x1 * s0) * INV_SQRT_HD);
        qsf[fr * 40 + j + 1] = f2bf_bits((x1 * c1 + x0 * s1) * INV_SQRT_HD);
        const unsigned short* kr = kbase + (size_t)fr * QKV_LD;
        x0 = bf2f(kr[j]); x1 = bf2f(kr[j + 1]);
        ksf[fr * 40 + j]     = f2bf_bits(x0 * c0 - x1 * s0);
        ksf[fr * 40 + j + 1] = f2bf_bits(x1 * c1 + x0 * s1);
        const unsigned short* vr = vtbase + (size_t)fr * vstride;
        vstf[j * 72 + fr]       = vr[j];
        vstf[(j + 1) * 72 + fr] = vr[j + 1];
    }
    __syncthreads();

    // ---- scores (16 q rows x 64 keys) ----
    v16bf qa = frag_A(qsf, wid * 16, 40);
    v8f S[4];
#pragma unroll
    for (int ni = 0; ni < 4; ++ni)
        S[ni] = wmma_bf16(qa, frag_B(ksf, ni * 16, 40), v8f_zero());

    // ---- full softmax over 64 keys ----
    float l[8];
#pragma unroll
    for (int r = 0; r < 8; ++r) {
        float mx = fmaxf(fmaxf(S[0][r], S[1][r]), fmaxf(S[2][r], S[3][r]));
        mx = half_max(mx);
        float rs = 0.f;
#pragma unroll
        for (int ni = 0; ni < 4; ++ni) {
            S[ni][r] = __expf(S[ni][r] - mx);
            rs += S[ni][r];
        }
        l[r] = half_sum(rs);
    }

    // ---- relayout P, then O = P @ V (K=64 in two chunks) ----
    unsigned short* pw = pb + wid * 16 * 72;
#pragma unroll
    for (int ni = 0; ni < 4; ++ni)
#pragma unroll
        for (int r = 0; r < 8; ++r)
            pw[(hf * 8 + r) * 72 + ni * 16 + (lane & 15)] = f2bf_bits(S[ni][r]);
    __syncthreads();

    v8f O[2] = { v8f_zero(), v8f_zero() };
#pragma unroll
    for (int kc = 0; kc < 2; ++kc) {
        v16bf pa = frag_A(pw + kc * 32, 0, 72);
        O[0] = wmma_bf16(pa, frag_B(vstf + kc * 32, 0, 72), O[0]);
        O[1] = wmma_bf16(pa, frag_B(vstf + kc * 32, 16, 72), O[1]);
    }

    const size_t ybase = (((size_t)b * T_LEN + t) * F_LEN) * DMODEL + h * HD;
#pragma unroll
    for (int ni = 0; ni < 2; ++ni)
#pragma unroll
        for (int r = 0; r < 8; ++r) {
            int fr = wid * 16 + hf * 8 + r;
            int d  = ni * 16 + (lane & 15);
            ybf[ybase + (size_t)fr * DMODEL + d] = f2bf_bits(O[ni][r] / l[r]);
        }
}

// ---------------------------------------------------------------------------
extern "C" void kernel_launch(void* const* d_in, const int* in_sizes, int n_in,
                              void* d_out, int out_size, void* d_ws, size_t ws_size,
                              hipStream_t stream)
{
    const float* x      = (const float*)d_in[0];   // (4,256,64,256)
    const float* W_attn = (const float*)d_in[1];   // (256,1280)
    const float* W_proj = (const float*)d_in[2];   // (256,256)
    const float* rot_t  = (const float*)d_in[3];   // (256,32)
    const float* rot_f  = (const float*)d_in[4];   // (64,32)

    unsigned short* qkv = (unsigned short*)d_ws;                    // 65536*1280 bf16
    unsigned short* vtb = qkv + (size_t)M_ROWS * QKV_LD;            // 65536*256 bf16
    unsigned short* ybf = vtb + (size_t)M_ROWS * DMODEL;            // 65536*256 bf16

    // 1) fused QKV projection: x(fp32) @ W_attn(fp32) -> qkv(bf16)
    dim3 g1(M_ROWS / 128, QKV_LD / 128);
    gemm_kernel<false, true><<<g1, 256, 0, stream>>>(x, W_attn, qkv,
                                                     M_ROWS, QKV_LD, DMODEL);
    // 2) time attention (rotary_t inside), -> vtb
    time_attn_kernel<<<B_SZ * F_LEN * HEADS, 256, 0, stream>>>(qkv, rot_t, vtb);
    // 3) frequency attention (rotary_f inside), -> ybf
    freq_attn_kernel<<<B_SZ * T_LEN * HEADS, 128, 0, stream>>>(qkv, rot_f, vtb, ybf);
    // 4) output projection: ybf(bf16) @ W_proj(fp32) -> d_out(fp32)
    dim3 g2(M_ROWS / 128, DMODEL / 128);
    gemm_kernel<true, false><<<g2, 256, 0, stream>>>(ybf, W_proj, d_out,
                                                     M_ROWS, DMODEL, DMODEL);
}
